// AttnBlock_13365938225576
// MI455X (gfx1250) — matrix operand order repaired and checked by reading, not compiled, not run on hardware
//
#include <hip/hip_runtime.h>
#include <hip/hip_bf16.h>

// ---------------------------------------------------------------------------
// AttnBlock (GroupNorm -> QKV -> softmax attention -> out-proj -> residual)
// B=4, C=256, H=W=64, N=4096, G=32. All GEMMs on v_wmma_f32_16x16x32_f16.
// Attention: 8 waves/block share async-staged K/V blocks in LDS (CDNA5
// 320KB-LDS + GLOBAL_LOAD_ASYNC_TO_LDS pipeline), cutting L2 traffic 8x.
// ---------------------------------------------------------------------------

#define B_   4
#define C_   256
#define N_   4096          // H*W
#define G_   32
#define CPG_ 8             // C/G
#define HW_  4096

typedef __attribute__((ext_vector_type(16))) _Float16 v16h;
typedef __attribute__((ext_vector_type(8)))  _Float16 v8h;
typedef __attribute__((ext_vector_type(8)))  float    v8f;
typedef __attribute__((ext_vector_type(4)))  int      v4i;

#if __has_builtin(__builtin_amdgcn_global_load_async_to_lds_b128) && \
    __has_builtin(__builtin_amdgcn_s_wait_asynccnt)
#define HAVE_ASYNC_LDS 1
#endif

// ---- stage 16 bytes (8 f16) global -> LDS ----------------------------------
__device__ __forceinline__ void stage16(const _Float16* g, _Float16* l) {
#ifdef HAVE_ASYNC_LDS
  __builtin_amdgcn_global_load_async_to_lds_b128(
      (__attribute__((address_space(1))) v4i*)g,
      (__attribute__((address_space(3))) v4i*)l, 0, 0);
#else
  *(v8h*)l = *(const v8h*)g;
#endif
}
__device__ __forceinline__ void stage_wait() {
#ifdef HAVE_ASYNC_LDS
  __builtin_amdgcn_s_wait_asynccnt(0);
#endif
}

// ---- WMMA wrapper: D = A(16x32 f16) * B(32x16 f16) + C(16x16 f32) ----------
__device__ __forceinline__ v8f wmma_f16(v16h a, v16h b, v8f c) {
  return __builtin_amdgcn_wmma_f32_16x16x32_f16(
      /*neg_a=*/false, a, /*neg_b=*/false, b,
      /*c_mod=*/(short)0, c, /*reuse_a=*/false, /*reuse_b=*/false);
}

// ---- A-tile loader: row-major source, A[m][k] = p[m*ld + k], 16x32 ---------
// ISA layout (16-bit A 16x32): lane L holds row M=L&15; lanes 0-15 hold
// K={0..7,16..23}, lanes 16-31 hold K={8..15,24..31}; 2 halves per VGPR.
__device__ __forceinline__ v16h load_a_tile(const _Float16* p, int ld) {
  const int lane = threadIdx.x & 31;
  const int m    = lane & 15;
  const int hi   = lane >> 4;
  const _Float16* row = p + (size_t)m * ld + hi * 8;
  v8h lo = *(const v8h*)(row);
  v8h hv = *(const v8h*)(row + 16);
  return __builtin_shufflevector(lo, hv, 0,1,2,3,4,5,6,7,8,9,10,11,12,13,14,15);
}

// ---- B-tile loader from "transposed" source: B[k][n] = p[n*ld + k], 32x16 --
// ISA layout (16-bit B 32x16): lane L holds column N=L&15; lanes 0-15 hold
// K={0..7,16..23}, lanes 16-31 hold K={8..15,24..31} (matches A tile).
__device__ __forceinline__ v16h load_b_tile(const _Float16* p, int ld) {
  const int lane = threadIdx.x & 31;
  const int n    = lane & 15;
  const int hi   = lane >> 4;
  const _Float16* src = p + (size_t)n * ld + hi * 8;
  v8h a = *(const v8h*)(src);
  v8h b = *(const v8h*)(src + 16);
  return __builtin_shufflevector(a, b, 0,1,2,3,4,5,6,7,8,9,10,11,12,13,14,15);
}

// ---------------------------------------------------------------------------
// Kernel 1: convert the four weight matrices f32 -> f16
// ---------------------------------------------------------------------------
__global__ void cvt_weights(const float* wq, const float* wk, const float* wv,
                            const float* wo, _Float16* wqh, _Float16* wkh,
                            _Float16* wvh, _Float16* woh) {
  int i = blockIdx.x * 256 + threadIdx.x;
  if (i < C_ * C_) {
    wqh[i] = (_Float16)wq[i];
    wkh[i] = (_Float16)wk[i];
    wvh[i] = (_Float16)wv[i];
    woh[i] = (_Float16)wo[i];
  }
}

// ---------------------------------------------------------------------------
// Kernel 2: GroupNorm. One block per (b,g); the (C/G)*HW slab is contiguous
// in NCHW. Writes hf[b][n][c] in f16.
// ---------------------------------------------------------------------------
__global__ __launch_bounds__(256) void gn_kernel(const float* __restrict__ x,
                                                 const float* __restrict__ gamma,
                                                 const float* __restrict__ beta,
                                                 _Float16* __restrict__ hf) {
  const int bg  = blockIdx.x;
  const int b   = bg >> 5;
  const int g   = bg & 31;
  const int tid = threadIdx.x;
  const float* base = x + ((size_t)b * C_ + (size_t)g * CPG_) * HW_;

  float s = 0.f, s2 = 0.f;
  for (int i = tid; i < CPG_ * HW_; i += 256) {
    float v = base[i];
    s += v; s2 += v * v;
  }
  __shared__ float r1[256], r2[256];
  r1[tid] = s; r2[tid] = s2;
  __syncthreads();
  for (int off = 128; off > 0; off >>= 1) {
    if (tid < off) { r1[tid] += r1[tid + off]; r2[tid] += r2[tid + off]; }
    __syncthreads();
  }
  const float inv_n = 1.0f / (float)(CPG_ * HW_);
  const float mean  = r1[0] * inv_n;
  const float var   = r2[0] * inv_n - mean * mean;
  const float rinv  = rsqrtf(var + 1e-5f);

  for (int i = tid; i < CPG_ * HW_; i += 256) {
    int cc = i >> 12;
    int n  = i & (HW_ - 1);
    int c  = g * CPG_ + cc;
    float v = (base[i] - mean) * rinv * gamma[c] + beta[c];
    hf[((size_t)b * N_ + n) * C_ + c] = (_Float16)v;
  }
}

// ---------------------------------------------------------------------------
// Kernel 3: QKV projections. Block = 8 waves sharing one staged A-tile
// (16 rows x 256) in LDS; each wave does one 16x16 output tile (8 WMMAs).
// q gets the c^-0.5 scale folded in; v is stored transposed [C][N].
// ---------------------------------------------------------------------------
__global__ __launch_bounds__(256) void qkv_kernel(
    const _Float16* __restrict__ hf,
    const _Float16* __restrict__ wqh, const _Float16* __restrict__ wkh,
    const _Float16* __restrict__ wvh,
    const float* __restrict__ bq, const float* __restrict__ bk,
    const float* __restrict__ bv,
    _Float16* __restrict__ q, _Float16* __restrict__ k,
    _Float16* __restrict__ vT) {
  __shared__ _Float16 ldsA[16 * C_];            // 8 KB shared A-tile

  int bid = blockIdx.x;                         // 3*4*256*2 blocks
  const int dhalf = bid & 1;   bid >>= 1;
  const int nt    = bid & 255; bid >>= 8;
  const int b     = bid & 3;   bid >>= 2;
  const int proj  = bid;                        // 0=q 1=k 2=v

  const _Float16* W    = (proj == 0) ? wqh : (proj == 1) ? wkh : wvh;
  const float*    bias = (proj == 0) ? bq  : (proj == 1) ? bk  : bv;

  const int wslot = threadIdx.x >> 5;
  const int lane  = threadIdx.x & 31;
  const int hi = lane >> 4, nn = lane & 15;
  const int n0 = nt * 16;
  const int dt = dhalf * 8 + wslot;
  const int d0 = dt * 16;

  // stage activation tile (16 contiguous rows of hf)
  const _Float16* A0 = hf + ((size_t)b * N_ + n0) * C_;
  for (int i = threadIdx.x; i < (16 * C_) / 8; i += 256)
    stage16(A0 + i * 8, &ldsA[i * 8]);
  stage_wait();
  __syncthreads();

  const float bval = bias[d0 + nn];
  v8f acc;
#pragma unroll
  for (int i = 0; i < 8; ++i) acc[i] = bval;

#pragma unroll
  for (int kk = 0; kk < 8; ++kk) {
    v16h a  = load_a_tile(&ldsA[kk * 32], C_);
    v16h bt = load_b_tile(W + (size_t)d0 * C_ + kk * 32, C_);
    acc = wmma_f16(a, bt, acc);
  }

  if (proj == 0) {
    const float scale = 0.0625f;                // 256^-0.5
#pragma unroll
    for (int r = 0; r < 8; ++r)
      q[((size_t)b * N_ + n0 + r + 8 * hi) * C_ + d0 + nn] =
          (_Float16)(acc[r] * scale);
  } else if (proj == 1) {
#pragma unroll
    for (int r = 0; r < 8; ++r)
      k[((size_t)b * N_ + n0 + r + 8 * hi) * C_ + d0 + nn] = (_Float16)acc[r];
  } else {
#pragma unroll
    for (int r = 0; r < 8; ++r)
      vT[((size_t)b * C_ + d0 + nn) * (size_t)N_ + n0 + r + 8 * hi] =
          (_Float16)acc[r];
  }
}

// ---------------------------------------------------------------------------
// Kernel 4: flash attention. 256 threads = 8 waves; each wave owns one
// 16-query tile (block covers 128 queries). K/V blocks (32 keys) are staged
// cooperatively in LDS with async-to-LDS and consumed by all 8 waves:
//   preload K0; loop: [issue V_j stage] scores(WMMA)+softmax -> P,
//   wait/barrier, [issue K_{j+1} stage] PV(WMMA), wait/barrier.
// LDS: 64KB Q + 16KB K + 16KB V + 8KB P = 104KB (CDNA5 320KB WGP LDS).
// ---------------------------------------------------------------------------
__global__ __launch_bounds__(256) void attn_kernel(
    const _Float16* __restrict__ q, const _Float16* __restrict__ k,
    const _Float16* __restrict__ vT, _Float16* __restrict__ attn) {
  __shared__ _Float16 ldsQ[8][16 * C_];   // 64 KB: per-wave q tile
  __shared__ _Float16 ldsK[32 * C_];      // 16 KB: K block, row-major [32][256]
  __shared__ _Float16 ldsV[C_ * 32];      // 16 KB: V block, [256][32]
  __shared__ _Float16 ldsP[8][16 * 32];   //  8 KB: per-wave P tile

  const int tid   = threadIdx.x;
  const int wslot = tid >> 5;
  const int lane  = tid & 31;
  const int hi = lane >> 4, nn = lane & 15;
  const int wid = blockIdx.x * 8 + wslot;       // 0..1023
  const int b   = wid >> 8;
  const int n0  = (wid & 255) << 4;

  const _Float16* qb = q  + ((size_t)b * N_ + n0) * C_;
  const _Float16* kb = k  + (size_t)b * N_ * C_;
  const _Float16* vb = vT + (size_t)b * C_ * (size_t)N_;

  // stage per-wave Q tile (contiguous 8KB) and cooperative K_0 (16KB)
  for (int i = lane; i < (16 * C_) / 8; i += 32)
    stage16(qb + i * 8, &ldsQ[wslot][i * 8]);
  for (int i = tid; i < (32 * C_) / 8; i += 256)
    stage16(kb + i * 8, &ldsK[i * 8]);
  stage_wait();
  __syncthreads();

  v8f O[16];
#pragma unroll
  for (int t = 0; t < 16; ++t)
#pragma unroll
    for (int i = 0; i < 8; ++i) O[t][i] = 0.f;
  float mrow[8], lrow[8];
#pragma unroll
  for (int r = 0; r < 8; ++r) { mrow[r] = -1e30f; lrow[r] = 0.f; }

  for (int j = 0; j < N_ / 32; ++j) {
    // ---- issue V_j staging: rows c=0..255, 32 f16 each (overlaps scores) --
#pragma unroll
    for (int i = tid; i < (C_ * 32) / 8; i += 256) {
      const int c = i >> 2, qd = i & 3;
      stage16(vb + (size_t)c * N_ + j * 32 + qd * 8, &ldsV[c * 32 + qd * 8]);
    }

    // ---- scores: S[16q x 32keys] = qTile @ K_j^T (scale folded into q) ----
    v8f s0, s1;
#pragma unroll
    for (int i = 0; i < 8; ++i) { s0[i] = 0.f; s1[i] = 0.f; }
#pragma unroll
    for (int kk = 0; kk < 8; ++kk) {
      v16h aq = load_a_tile(&ldsQ[wslot][kk * 32], C_);
      v16h b0 = load_b_tile(&ldsK[kk * 32], C_);
      v16h b1 = load_b_tile(&ldsK[16 * C_ + kk * 32], C_);
      s0 = wmma_f16(aq, b0, s0);
      s1 = wmma_f16(aq, b1, s1);
    }

    // ---- online softmax; row r of each half lives at VGPR r --------------
    float corr[8];
#pragma unroll
    for (int r = 0; r < 8; ++r) {
      float mx = fmaxf(s0[r], s1[r]);
#pragma unroll
      for (int msk = 1; msk < 16; msk <<= 1)
        mx = fmaxf(mx, __shfl_xor(mx, msk, 32));
      const float mnew = fmaxf(mrow[r], mx);
      corr[r] = __expf(mrow[r] - mnew);
      mrow[r] = mnew;
      const float p0 = __expf(s0[r] - mnew);
      const float p1 = __expf(s1[r] - mnew);
      s0[r] = p0; s1[r] = p1;
      float sum = p0 + p1;
#pragma unroll
      for (int msk = 1; msk < 16; msk <<= 1)
        sum += __shfl_xor(sum, msk, 32);
      lrow[r] = lrow[r] * corr[r] + sum;
    }
#pragma unroll
    for (int t = 0; t < 16; ++t)
#pragma unroll
      for (int r = 0; r < 8; ++r) O[t][r] *= corr[r];

    // ---- P (D layout) -> row-major per-wave LDS tile ---------------------
#pragma unroll
    for (int r = 0; r < 8; ++r) {
      const int m = r + 8 * hi;
      ldsP[wslot][m * 32 + nn]      = (_Float16)s0[r];
      ldsP[wslot][m * 32 + nn + 16] = (_Float16)s1[r];
    }

    stage_wait();          // V_j landed
    __syncthreads();       // all waves done reading K_j, P visible

    // ---- issue K_{j+1} staging (overlaps PV WMMAs) -----------------------
    if (j + 1 < N_ / 32) {
      const _Float16* kn = kb + (size_t)(j + 1) * 32 * C_;
#pragma unroll
      for (int i = tid; i < (32 * C_) / 8; i += 256)
        stage16(kn + i * 8, &ldsK[i * 8]);
    }

    // ---- O += P @ V_j ----------------------------------------------------
    v16h ap = load_a_tile(&ldsP[wslot][0], 32);
#pragma unroll
    for (int t = 0; t < 16; ++t) {
      v16h bv = load_b_tile(&ldsV[(size_t)t * 16 * 32], 32);
      O[t] = wmma_f16(ap, bv, O[t]);
    }

    stage_wait();          // K_{j+1} landed
    __syncthreads();       // all waves done reading V_j
  }

#pragma unroll
  for (int r = 0; r < 8; ++r) lrow[r] = 1.0f / lrow[r];
#pragma unroll
  for (int t = 0; t < 16; ++t)
#pragma unroll
    for (int r = 0; r < 8; ++r)
      attn[((size_t)b * N_ + n0 + r + 8 * hi) * C_ + t * 16 + nn] =
          (_Float16)(O[t][r] * lrow[r]);
}

// ---------------------------------------------------------------------------
// Kernel 5: out-projection + bias + residual, f32 NCHW output.
// Block = 8 waves sharing one staged attn A-tile in LDS.
// ---------------------------------------------------------------------------
__global__ __launch_bounds__(256) void out_kernel(
    const _Float16* __restrict__ attn, const _Float16* __restrict__ woh,
    const float* __restrict__ bo, const float* __restrict__ x,
    float* __restrict__ out) {
  __shared__ _Float16 ldsA[16 * C_];            // 8 KB

  int bid = blockIdx.x;                         // 4*256*2 blocks
  const int dhalf = bid & 1;   bid >>= 1;
  const int nt    = bid & 255; bid >>= 8;
  const int b     = bid;

  const int wslot = threadIdx.x >> 5;
  const int lane  = threadIdx.x & 31;
  const int hi = lane >> 4, nn = lane & 15;
  const int n0 = nt * 16;
  const int d0 = (dhalf * 8 + wslot) * 16;

  const _Float16* A0 = attn + ((size_t)b * N_ + n0) * C_;
  for (int i = threadIdx.x; i < (16 * C_) / 8; i += 256)
    stage16(A0 + i * 8, &ldsA[i * 8]);
  stage_wait();
  __syncthreads();

  const float bval = bo[d0 + nn];
  v8f acc;
#pragma unroll
  for (int i = 0; i < 8; ++i) acc[i] = bval;

#pragma unroll
  for (int kk = 0; kk < 8; ++kk) {
    v16h a  = load_a_tile(&ldsA[kk * 32], C_);
    v16h bt = load_b_tile(woh + (size_t)d0 * C_ + kk * 32, C_);
    acc = wmma_f16(a, bt, acc);
  }

#pragma unroll
  for (int r = 0; r < 8; ++r) {
    size_t idx = ((size_t)b * C_ + d0 + nn) * (size_t)N_ + n0 + r + 8 * hi;
    out[idx] = x[idx] + acc[r];
  }
}

// ---------------------------------------------------------------------------
// Launch
// ---------------------------------------------------------------------------
extern "C" void kernel_launch(void* const* d_in, const int* in_sizes, int n_in,
                              void* d_out, int out_size, void* d_ws, size_t ws_size,
                              hipStream_t stream) {
  const float* x     = (const float*)d_in[0];
  const float* gamma = (const float*)d_in[1];
  const float* beta  = (const float*)d_in[2];
  const float* wq    = (const float*)d_in[3];
  const float* bq    = (const float*)d_in[4];
  const float* wk    = (const float*)d_in[5];
  const float* bk    = (const float*)d_in[6];
  const float* wv    = (const float*)d_in[7];
  const float* bv    = (const float*)d_in[8];
  const float* wo    = (const float*)d_in[9];
  const float* bo    = (const float*)d_in[10];
  float* out = (float*)d_out;

  char* ws = (char*)d_ws;
  const size_t act_bytes = (size_t)B_ * N_ * C_ * sizeof(_Float16); // 8 MB
  const size_t w_bytes   = (size_t)C_ * C_ * sizeof(_Float16);      // 128 KB
  _Float16* hf   = (_Float16*)(ws + 0 * act_bytes);
  _Float16* q    = (_Float16*)(ws + 1 * act_bytes);
  _Float16* k    = (_Float16*)(ws + 2 * act_bytes);
  _Float16* vT   = (_Float16*)(ws + 3 * act_bytes);
  _Float16* attn = (_Float16*)(ws + 4 * act_bytes);
  _Float16* wqh  = (_Float16*)(ws + 5 * act_bytes + 0 * w_bytes);
  _Float16* wkh  = (_Float16*)(ws + 5 * act_bytes + 1 * w_bytes);
  _Float16* wvh  = (_Float16*)(ws + 5 * act_bytes + 2 * w_bytes);
  _Float16* woh  = (_Float16*)(ws + 5 * act_bytes + 3 * w_bytes);

  cvt_weights<<<(C_ * C_ + 255) / 256, 256, 0, stream>>>(
      wq, wk, wv, wo, wqh, wkh, wvh, woh);

  gn_kernel<<<B_ * G_, 256, 0, stream>>>(x, gamma, beta, hf);

  // 3*4*256*2 = 6144 blocks, 8 waves each (one 16x16 tile per wave)
  qkv_kernel<<<3 * B_ * (N_ / 16) * 2, 256, 0, stream>>>(
      hf, wqh, wkh, wvh, bq, bk, bv, q, k, vT);

  // 1024 query tiles / 8 waves per block = 128 blocks
  attn_kernel<<<(B_ * (N_ / 16)) / 8, 256, 0, stream>>>(q, k, vT, attn);

  // 4*256*2 = 2048 blocks
  out_kernel<<<B_ * (N_ / 16) * 2, 256, 0, stream>>>(attn, woh, bo, x, out);
}
